// CoAttention_69664369541627
// MI455X (gfx1250) — compile-verified, matrix operand order checked by reading, not run on hardware
//
#include <hip/hip_runtime.h>
#include <hip/hip_bf16.h>
#include <math.h>

// ---------------------------------------------------------------------------
// CoAttention for MI455X (gfx1250, wave32).
// All matmuls use V_WMMA_F32_16X16X32_BF16 (f32 accum, bf16 operands converted
// at LDS-staging time; HBM traffic stays f32).  LV=576 padded to 640 with
// exact-zero affinity pad columns so padded regions contribute nothing.
// Workspace requirement: ~480 MB of f32 scratch in d_ws.
// ---------------------------------------------------------------------------

typedef __bf16 bf16;
typedef __bf16 v16bf __attribute__((ext_vector_type(16)));
typedef __bf16 v8bf  __attribute__((ext_vector_type(8)));
typedef __bf16 v4bf  __attribute__((ext_vector_type(4)));
typedef float  v8f   __attribute__((ext_vector_type(8)));
typedef float  v4f   __attribute__((ext_vector_type(4)));

#define BM 128
#define BN 128
#define BK 32
#define LDSP 56   // LDS row pitch (elements): 112B rows, 16B aligned, bank-conflict-free

// ---------------------------------------------------------------------------
// Generic tiled bf16-WMMA GEMM:  C[M,N] = A[M,K] * B[K,N]   (f32 in/out)
//   TRANS_A: logical A[m][k] = Aphys[k*lda + m]   (used for aff^T @ wq_q)
//   TRANS_B: logical B[k][n] = Bphys[n*ldb + k]   (used for TB @ img^T)
// Mreal/Nreal: real extents; staging zero-fills beyond them (pad -> exact 0).
// Grid: (N/BN, M/BM, batch).  Block: 256 threads = 8 waves of 32.
// ---------------------------------------------------------------------------
template<bool TRANS_A, bool TRANS_B>
__global__ __launch_bounds__(256) void gemm_bf16_wmma(
    const float* __restrict__ A, const float* __restrict__ B, float* __restrict__ C,
    int M, int N, int K, int lda, int ldb, int ldc,
    long long strideA, long long strideB, long long strideC,
    int Mreal, int Nreal)
{
  (void)M; (void)N;
  __shared__ __align__(16) bf16 As[BM * LDSP];   // As[m][k]
  __shared__ __align__(16) bf16 Bs[BN * LDSP];   // Bs[n][k]  (transposed in LDS)

  const int tid = threadIdx.x;
  A += (long long)blockIdx.z * strideA;
  B += (long long)blockIdx.z * strideB;
  C += (long long)blockIdx.z * strideC;

  const int m0 = blockIdx.y * BM;
  const int n0 = blockIdx.x * BN;
  const int w  = tid >> 5, l = tid & 31;
  const int wm = w & 1;        // 64-row half
  const int wn = w >> 1;       // 32-col quarter
  const int lrow  = l & 15;
  const int lhalf = l >> 4;
  const int koff  = lhalf * 8; // K sub-offset per 16-bit WMMA A/B layout

  v8f acc[4][2];
  const v8f vzero = {0.f,0.f,0.f,0.f,0.f,0.f,0.f,0.f};
#pragma unroll
  for (int mi = 0; mi < 4; ++mi)
#pragma unroll
    for (int ni = 0; ni < 2; ++ni) acc[mi][ni] = vzero;

  const int nkb = K / BK;
  for (int kb = 0; kb < nkb; ++kb) {
    const int k0 = kb * BK;

    // ---- stage A tile -> As[m][k] (f32 -> bf16) ----
    if (!TRANS_A) {
      const int kk = (tid & 7) * 4;
#pragma unroll
      for (int p = 0; p < 4; ++p) {
        const int mr = (tid >> 3) + p * 32;
        const int gm = m0 + mr;
        v4f v = {0.f,0.f,0.f,0.f};
        if (gm < Mreal)
          v = *reinterpret_cast<const v4f*>(A + (long long)gm * lda + (k0 + kk));
        v4bf h;
#pragma unroll
        for (int j = 0; j < 4; ++j) h[j] = (bf16)v[j];
        *reinterpret_cast<v4bf*>(&As[mr * LDSP + kk]) = h;
      }
      if (kb + 1 < nkb)  // speculative hint for next K-slab -> global_prefetch_b8
        __builtin_prefetch(A + (long long)(m0 + (tid >> 3)) * lda + (k0 + BK + kk), 0, 1);
    } else {
      const int m4 = (tid & 31) * 4;
#pragma unroll
      for (int p = 0; p < 4; ++p) {
        const int kk = (tid >> 5) + p * 8;
        v4f v = {0.f,0.f,0.f,0.f};
        if (m0 + m4 < Mreal)
          v = *reinterpret_cast<const v4f*>(A + (long long)(k0 + kk) * lda + (m0 + m4));
#pragma unroll
        for (int j = 0; j < 4; ++j) As[(m4 + j) * LDSP + kk] = (bf16)v[j];
      }
    }

    // ---- stage B tile -> Bs[n][k] ----
    if (!TRANS_B) {
      const int n4 = (tid & 31) * 4;
#pragma unroll
      for (int p = 0; p < 4; ++p) {
        const int kk = (tid >> 5) + p * 8;
        v4f v = {0.f,0.f,0.f,0.f};
        if (n0 + n4 < Nreal)
          v = *reinterpret_cast<const v4f*>(B + (long long)(k0 + kk) * ldb + (n0 + n4));
#pragma unroll
        for (int j = 0; j < 4; ++j) Bs[(n4 + j) * LDSP + kk] = (bf16)v[j];
      }
    } else {
      const int kk = (tid & 7) * 4;
#pragma unroll
      for (int p = 0; p < 4; ++p) {
        const int nr = (tid >> 3) + p * 32;
        const int gn = n0 + nr;
        v4f v = {0.f,0.f,0.f,0.f};
        if (gn < Nreal)
          v = *reinterpret_cast<const v4f*>(B + (long long)gn * ldb + (k0 + kk));
        v4bf h;
#pragma unroll
        for (int j = 0; j < 4; ++j) h[j] = (bf16)v[j];
        *reinterpret_cast<v4bf*>(&Bs[nr * LDSP + kk]) = h;
      }
    }
    __syncthreads();

    // ---- gather fragments (two 16B ds reads each) and WMMA ----
    v16bf afr[4], bfr[2];
#pragma unroll
    for (int mi = 0; mi < 4; ++mi) {
      const bf16* base = &As[(wm * 64 + mi * 16 + lrow) * LDSP];
      v8bf lo = *reinterpret_cast<const v8bf*>(base + koff);
      v8bf hi = *reinterpret_cast<const v8bf*>(base + 16 + koff);
      afr[mi] = __builtin_shufflevector(lo, hi, 0,1,2,3,4,5,6,7,8,9,10,11,12,13,14,15);
    }
#pragma unroll
    for (int ni = 0; ni < 2; ++ni) {
      const bf16* base = &Bs[(wn * 32 + ni * 16 + lrow) * LDSP];
      v8bf lo = *reinterpret_cast<const v8bf*>(base + koff);
      v8bf hi = *reinterpret_cast<const v8bf*>(base + 16 + koff);
      bfr[ni] = __builtin_shufflevector(lo, hi, 0,1,2,3,4,5,6,7,8,9,10,11,12,13,14,15);
    }
#pragma unroll
    for (int mi = 0; mi < 4; ++mi)
#pragma unroll
      for (int ni = 0; ni < 2; ++ni)
        acc[mi][ni] = __builtin_amdgcn_wmma_f32_16x16x32_bf16(
            false, afr[mi], false, bfr[ni], (short)0, acc[mi][ni], false, false);
    __syncthreads();
  }

  // ---- epilogue: 16x16 f32 D layout (lane n = l%16, rows vg + 8*(l/16)) ----
#pragma unroll
  for (int mi = 0; mi < 4; ++mi)
#pragma unroll
    for (int ni = 0; ni < 2; ++ni) {
      const int n  = n0 + wn * 32 + ni * 16 + lrow;
      const int mb = m0 + wm * 64 + mi * 16 + lhalf * 8;
#pragma unroll
      for (int vg = 0; vg < 8; ++vg)
        C[(long long)(mb + vg) * ldc + n] = acc[mi][ni][vg];
    }
}

// ---------------------------------------------------------------------------
// s[b,i] = sum_k tanh(x1[b,i,k] + x2[b,i,k]) * wh[k]     (K = 128)
// grid = (Lreal, B), block = 128.  pitch = padded row count of x1/x2.
// ---------------------------------------------------------------------------
__global__ __launch_bounds__(128) void score_kernel(
    const float* __restrict__ x1, const float* __restrict__ x2,
    const float* __restrict__ wh, float* __restrict__ s, int pitch, int Lreal)
{
  __shared__ float red[128];
  const int i = blockIdx.x, b = blockIdx.y, k = threadIdx.x;
  const long long idx = ((long long)b * pitch + i) * 128 + k;
  red[k] = tanhf(x1[idx] + x2[idx]) * wh[k];
  __syncthreads();
  for (int off = 64; off > 0; off >>= 1) {
    if (k < off) red[k] += red[k + off];
    __syncthreads();
  }
  if (k == 0) s[(long long)b * Lreal + i] = red[0];
}

// softmax over L entries per batch row.  grid = B, block = 256.
__global__ __launch_bounds__(256) void softmax_kernel(
    const float* __restrict__ s, float* __restrict__ a, int L)
{
  __shared__ float red[256];
  const int b = blockIdx.x, t = threadIdx.x;
  const float* sb = s + (long long)b * L;
  float m = -1e30f;
  for (int i = t; i < L; i += 256) m = fmaxf(m, sb[i]);
  red[t] = m; __syncthreads();
  for (int off = 128; off > 0; off >>= 1) {
    if (t < off) red[t] = fmaxf(red[t], red[t + off]);
    __syncthreads();
  }
  m = red[0]; __syncthreads();
  float sum = 0.f;
  for (int i = t; i < L; i += 256) sum += __expf(sb[i] - m);
  red[t] = sum; __syncthreads();
  for (int off = 128; off > 0; off >>= 1) {
    if (t < off) red[t] += red[t + off];
    __syncthreads();
  }
  const float inv = 1.f / red[0];
  for (int i = t; i < L; i += 256) a[(long long)b * L + i] = __expf(sb[i] - m) * inv;
}

// ctx[b,e] = sum_y av[b,y]*img[b,y,e] + sum_x aq[b,x]*txt[b,x,e]
// grid = (E/256, B), block = 256.  attn weights are block-uniform -> s_loads.
__global__ __launch_bounds__(256) void context_kernel(
    const float* __restrict__ av, const float* __restrict__ aq,
    const float* __restrict__ img, const float* __restrict__ txt,
    float* __restrict__ ctx)
{
  const int b = blockIdx.y;
  const int e = blockIdx.x * 256 + threadIdx.x;
  const float* ib  = img + (long long)b * 576 * 768;
  const float* tb  = txt + (long long)b * 1024 * 768;
  const float* avb = av + (long long)b * 576;
  const float* aqb = aq + (long long)b * 1024;
  float acc = 0.f;
  for (int y = 0; y < 576; ++y)  acc += avb[y] * ib[(long long)y * 768 + e];
  for (int x = 0; x < 1024; ++x) acc += aqb[x] * tb[(long long)x * 768 + e];
  ctx[(long long)b * 768 + e] = acc;
}

// out[b,n] = tanh(sum_e ctx[b,e]*w_s[e,n]).  grid = (E/256, B). (77 MFLOP total)
__global__ __launch_bounds__(256) void final_kernel(
    const float* __restrict__ ctx, const float* __restrict__ wS,
    float* __restrict__ out)
{
  const int b = blockIdx.y;
  const int n = blockIdx.x * 256 + threadIdx.x;
  const float* cb = ctx + (long long)b * 768;
  float acc = 0.f;
  for (int e = 0; e < 768; ++e) acc += cb[e] * wS[(long long)e * 768 + n];
  out[(long long)b * 768 + n] = tanhf(acc);
}

// ---------------------------------------------------------------------------
extern "C" void kernel_launch(void* const* d_in, const int* in_sizes, int n_in,
                              void* d_out, int out_size, void* d_ws, size_t ws_size,
                              hipStream_t stream)
{
  (void)in_sizes; (void)n_in; (void)out_size; (void)ws_size;
  const float* txt  = (const float*)d_in[0];   // [64,1024,768]
  const float* img  = (const float*)d_in[1];   // [64, 576,768]
  /* d_in[2] = text_attention_mask (int64) — unused by the reference */
  const float* w_b  = (const float*)d_in[3];   // [768,768]
  const float* w_v  = (const float*)d_in[4];   // [768,128]
  const float* w_q  = (const float*)d_in[5];   // [768,128]
  const float* w_hv = (const float*)d_in[6];   // [128]
  const float* w_hq = (const float*)d_in[7];   // [128]
  const float* w_s  = (const float*)d_in[8];   // [768,768]
  float* out = (float*)d_out;                  // [64,768] f32

  float* ws = (float*)d_ws;
  size_t off = 0;
  float* TB   = ws + off; off += (size_t)64 * 1024 * 768;  // text @ w_b
  float* AFF  = ws + off; off += (size_t)64 * 1024 * 640;  // affinity (LV padded to 640)
  float* WQQ  = ws + off; off += (size_t)64 * 1024 * 128;  // text @ w_q
  float* WVV  = ws + off; off += (size_t)64 *  640 * 128;  // img @ w_v (padded rows = 0)
  float* WQQC = ws + off; off += (size_t)64 *  640 * 128;  // aff^T @ wq_q
  float* WVVC = ws + off; off += (size_t)64 * 1024 * 128;  // aff @ wv_v
  float* SV   = ws + off; off += (size_t)64 * 576;
  float* SQ   = ws + off; off += (size_t)64 * 1024;
  float* AV   = ws + off; off += (size_t)64 * 576;
  float* AQ   = ws + off; off += (size_t)64 * 1024;
  float* CTX  = ws + off; off += (size_t)64 * 768;

  // 1) TB = text @ w_b                 [65536 x 768] x [768 x 768]
  gemm_bf16_wmma<false,false><<<dim3(6,512,1),256,0,stream>>>(
      txt, w_b, TB, 65536,768,768, 768,768,768, 0,0,0, 65536,768);
  // 2) WQQ = text @ w_q                [65536 x 128]
  gemm_bf16_wmma<false,false><<<dim3(1,512,1),256,0,stream>>>(
      txt, w_q, WQQ, 65536,128,768, 768,128,128, 0,0,0, 65536,128);
  // 3) WVV[b] = img[b] @ w_v           batched, M 576->640 (pad rows -> 0)
  gemm_bf16_wmma<false,false><<<dim3(1,5,64),256,0,stream>>>(
      img, w_v, WVV, 640,128,768, 768,128,128,
      (long long)576*768, 0, (long long)640*128, 576,128);
  // 4) AFF[b] = TB[b] @ img[b]^T       batched NT, N 576->640 (pad cols -> 0)
  gemm_bf16_wmma<false,true><<<dim3(5,8,64),256,0,stream>>>(
      TB, img, AFF, 1024,640,768, 768,768,640,
      (long long)1024*768, (long long)576*768, (long long)1024*640, 1024,576);
  // 5) WVVC[b] = AFF[b] @ WVV[b]       (zero aff pad cols kill WVV pad rows)
  gemm_bf16_wmma<false,false><<<dim3(1,8,64),256,0,stream>>>(
      AFF, WVV, WVVC, 1024,128,640, 640,128,128,
      (long long)1024*640, (long long)640*128, (long long)1024*128, 1024,128);
  // 6) WQQC[b] = AFF[b]^T @ WQQ[b]     batched TN
  gemm_bf16_wmma<true,false><<<dim3(1,5,64),256,0,stream>>>(
      AFF, WQQ, WQQC, 640,128,1024, 640,128,128,
      (long long)1024*640, (long long)1024*128, (long long)640*128, 640,128);
  // 7) scores: s = (tanh(x1+x2)) . w_h
  score_kernel<<<dim3(576,64),128,0,stream>>>(WVV, WQQC, w_hv, SV, 640, 576);
  score_kernel<<<dim3(1024,64),128,0,stream>>>(WQQ, WVVC, w_hq, SQ, 1024, 1024);
  // 8) softmax per batch row
  softmax_kernel<<<64,256,0,stream>>>(SV, AV, 576);
  softmax_kernel<<<64,256,0,stream>>>(SQ, AQ, 1024);
  // 9) ctx = attn_v @ img + attn_q @ txt
  context_kernel<<<dim3(3,64),256,0,stream>>>(AV, AQ, img, txt, CTX);
  // 10) out = tanh(ctx @ w_s)
  final_kernel<<<dim3(3,64),256,0,stream>>>(CTX, w_s, out);
}
